// IRCA_2018634629362
// MI455X (gfx1250) — matrix operand order repaired and checked by reading, not compiled, not run on hardware
//
#include <hip/hip_runtime.h>

// ---------------------------------------------------------------------------
// Types for CDNA5 WMMA (wave32): V_WMMA_F32_16X16X32_BF16
// ---------------------------------------------------------------------------
typedef __bf16 bf16_t;
typedef bf16_t v16bf __attribute__((ext_vector_type(16)));
typedef float  v8f   __attribute__((ext_vector_type(8)));
typedef unsigned int v4u __attribute__((ext_vector_type(4)));
typedef int v8i __attribute__((ext_vector_type(8)));
typedef int v4i __attribute__((ext_vector_type(4)));

union Frag {
    uint4 u[2];   // two 16-byte loads
    v16bf v;      // 16 bf16 elements = 8 VGPRs
};

__device__ __forceinline__ unsigned short f32_to_bf16(float f) {
    unsigned int u = __float_as_uint(f);
    unsigned int r = u + 0x7FFFu + ((u >> 16) & 1u);  // round-to-nearest-even
    return (unsigned short)(r >> 16);
}
__device__ __forceinline__ float bf16_to_f32(unsigned short h) {
    return __uint_as_float(((unsigned int)h) << 16);
}

#define WMMA_BF16(A, B, C) \
    __builtin_amdgcn_wmma_f32_16x16x32_bf16(false, (A), false, (B), (short)0, (C), false, false)

// ---------------------------------------------------------------------------
// TDM: DMA one 16-row x 512-byte tile (bf16 means slice) from global to LDS,
// padding each 512B row by 16B so the LDS image has a 528B (264-elem) stride.
// D# packing per CDNA5 ISA:
//   group0: count=1 | lds_addr | global_addr[56:0] | type=2
//   group1: data_size=4B, pad_enable, pad_interval=6 (128 DW = 512B),
//           pad_amount=3 (4 DW = 16B), tensor_dim0=128, tensor_dim1=256,
//           tile_dim0=128, tile_dim1=16, tensor_dim0_stride=128
// Tracked on TENSORcnt; EXEC is ignored (one issue per wave).
// This toolchain exposes the 6-arg builtin: (g0, g1, g2, g3, extra, cpol).
// ---------------------------------------------------------------------------
__device__ __forceinline__ void tdm_load_tile16(const unsigned short* gsrc,
                                                void* lds_dst)
{
    const unsigned long long ga = (unsigned long long)(size_t)gsrc;

    v4u g0;
    g0.x = 1u;                                   // count=1 (valid user D#)
    g0.y = (unsigned int)(size_t)lds_dst;        // LDS byte address
    g0.z = (unsigned int)ga;                     // global_addr[31:0]
    g0.w = (unsigned int)((ga >> 32) & 0x01FFFFFFu) | (2u << 30);  // [56:32]|type=2

    v8i g1;
    g1[0] = (int)((2u << 16)      // data_size = 4 bytes
                | (1u << 20)      // pad_enable
                | (6u << 22)      // pad_interval: 2^(6+1)=128 DWORDs = 512B
                | (3u << 25));    // pad_amount: 4 DWORDs = 16B
    g1[1] = (int)(128u << 16);    // tensor_dim0 = 128 (4B units)
    g1[2] = (int)(256u << 16);    // tensor_dim1 = 256 rows
    g1[3] = (int)(128u << 16);    // tile_dim0 = 128 (512B row)
    g1[4] = 16;                   // tile_dim1 = 16 rows, tile_dim2 = 0
    g1[5] = 128;                  // tensor_dim0_stride = 128 (contiguous rows)
    g1[6] = 0;
    g1[7] = 0;

    const v4i gz  = {0, 0, 0, 0};              // groups 2/3 unused (2D tile)
    const v8i gz8 = {0, 0, 0, 0, 0, 0, 0, 0};  // extra group (zero-filled)
    __builtin_amdgcn_tensor_load_to_lds(g0, g1, gz, gz, gz8, 0);
}

// ---------------------------------------------------------------------------
// Kernel 1: per-row L2 normalize (c=256 rows, d=256) + bf16 hi/lo split
// ---------------------------------------------------------------------------
__global__ void __launch_bounds__(256)
row_l2norm_split_kernel(const float* __restrict__ src,
                        float* __restrict__ outf,
                        unsigned short* __restrict__ oh,
                        unsigned short* __restrict__ ol)
{
    const int c = blockIdx.x;
    const int t = threadIdx.x;
    const float v = src[c * 256 + t];
    float ss = v * v;
#pragma unroll
    for (int off = 16; off >= 1; off >>= 1) ss += __shfl_xor(ss, off, 32);
    __shared__ float red[8];
    if ((t & 31) == 0) red[t >> 5] = ss;
    __syncthreads();
    float tot = 0.f;
#pragma unroll
    for (int i = 0; i < 8; ++i) tot += red[i];
    const float inv = 1.0f / fmaxf(sqrtf(tot), 1e-12f);
    const float xn = v * inv;
    outf[c * 256 + t] = xn;
    const unsigned short h = f32_to_bf16(xn);
    oh[c * 256 + t] = h;
    ol[c * 256 + t] = f32_to_bf16(xn - bf16_to_f32(h));
}

// ---------------------------------------------------------------------------
// Kernel 1b: plain bf16 hi/lo split (Wk, Wv)
// ---------------------------------------------------------------------------
__global__ void __launch_bounds__(256)
split_kernel(const float* __restrict__ src,
             unsigned short* __restrict__ oh,
             unsigned short* __restrict__ ol, int n)
{
    const int i = blockIdx.x * 256 + threadIdx.x;
    if (i < n) {
        const float v = src[i];
        const unsigned short h = f32_to_bf16(v);
        oh[i] = h;
        ol[i] = f32_to_bf16(v - bf16_to_f32(h));
    }
}

// ---------------------------------------------------------------------------
// Kernel 2 (main, fused): block = 128 threads = 4 waves, each wave owns a
// 16-row tile of x. Per block:
//   - wave0 TDM-prefetches the nt=0 means slice (hi+lo) into LDS
//   - each wave loads + L2-normalizes its tile, splits to bf16 hi/lo in LDS
//   - block-synced loop over 16 N-tiles: wave0 TDM-prefetches slice nt+1
//     (double buffered) while all waves run 24 split-bf16 WMMAs against the
//     staged slice; per-lane running argmax in C-fragment registers
//   - xor-shuffle argmax reduction; scatter-add into sums/bins (f32 atomics)
// Grid = 1024 blocks -> 65536 rows. LDS ~99KB/block (2 blocks/WGP of 320KB).
// ---------------------------------------------------------------------------
__global__ void __launch_bounds__(128)
irca_dist_kernel(const float* __restrict__ x,
                 const unsigned short* __restrict__ mh,
                 const unsigned short* __restrict__ ml,
                 float* __restrict__ sums,
                 float* __restrict__ bins)
{
    // padded stride 264 elems (528 B): 16B-aligned, minimal-conflict b128 reads
    __shared__ __align__(16) unsigned short sh_hi[4][16][264];
    __shared__ __align__(16) unsigned short sh_lo[4][16][264];
    __shared__ __align__(16) unsigned short mb_hi[2][16][264];  // staged means
    __shared__ __align__(16) unsigned short mb_lo[2][16][264];

    const int lane = threadIdx.x & 31;
    const int wv   = threadIdx.x >> 5;
    const int tile = blockIdx.x * 4 + wv;
    const int row0 = tile * 16;

    // Prefetch means slice for nt=0 via TDM (wave0 only; EXEC-independent)
    if (wv == 0) {
        tdm_load_tile16(mh, &mb_hi[0][0][0]);
        tdm_load_tile16(ml, &mb_lo[0][0][0]);
    }

    // ---- Phase 1: load + normalize (2 lanes per row, 128 elems each) ----
    const int r   = lane >> 1;
    const int seg = lane & 1;
    const float4* src = (const float4*)(x + (size_t)(row0 + r) * 256 + seg * 128);

    float4 buf[32];
    float ss = 0.f;
#pragma unroll
    for (int i = 0; i < 32; ++i) {
        buf[i] = src[i];
        ss += buf[i].x * buf[i].x + buf[i].y * buf[i].y +
              buf[i].z * buf[i].z + buf[i].w * buf[i].w;
    }
    ss += __shfl_xor(ss, 1, 32);
    const float inv = 1.0f / fmaxf(sqrtf(ss), 1e-12f);

    unsigned short* dsth = &sh_hi[wv][r][seg * 128];
    unsigned short* dstl = &sh_lo[wv][r][seg * 128];
#pragma unroll
    for (int i = 0; i < 32; ++i) {
        const float vals[4] = {buf[i].x, buf[i].y, buf[i].z, buf[i].w};
#pragma unroll
        for (int j = 0; j < 4; ++j) {
            const float xn = vals[j] * inv;
            const unsigned short h = f32_to_bf16(xn);
            dsth[i * 4 + j] = h;
            dstl[i * 4 + j] = f32_to_bf16(xn - bf16_to_f32(h));
        }
    }
    if (wv == 0) __builtin_amdgcn_s_wait_tensorcnt(0);
    __syncthreads();

    // ---- Phase 2: WMMA dists + argmax ----
    const int m  = lane & 15;
    const int hh = lane >> 4;

    // A-fragment (16-bit, 16x32): lane row = l&15; elems 0..7 at col 32kb+8h,
    // elems 8..15 at col 32kb+16+8h (row-major contiguous -> b128 LDS loads)
    const unsigned short* arow_h = &sh_hi[wv][m][0];
    const unsigned short* arow_l = &sh_lo[wv][m][0];
    Frag Ah[8], Al[8];
#pragma unroll
    for (int kb = 0; kb < 8; ++kb) {
        const int c0 = 32 * kb + 8 * hh;
        Ah[kb].u[0] = *(const uint4*)(arow_h + c0);
        Ah[kb].u[1] = *(const uint4*)(arow_h + c0 + 16);
        Al[kb].u[0] = *(const uint4*)(arow_l + c0);
        Al[kb].u[1] = *(const uint4*)(arow_l + c0 + 16);
    }

    float best[8];
    int   bidx[8];
#pragma unroll
    for (int i = 0; i < 8; ++i) { best[i] = -3.4e38f; bidx[i] = 0; }

    for (int nt = 0; nt < 16; ++nt) {
        // Prefetch next means slice into the other buffer while computing
        if (wv == 0 && nt < 15) {
            const size_t off = (size_t)(nt + 1) * 16 * 256;
            tdm_load_tile16(mh + off, &mb_hi[(nt + 1) & 1][0][0]);
            tdm_load_tile16(ml + off, &mb_lo[(nt + 1) & 1][0][0]);
        }

        const int cidx = nt * 16 + m;  // this lane's cluster column
        // B-fragment: lane N = l&15 -> staged row m; elems 0..7 at K 32kb+8h, 8..15 at K 32kb+16+8h
        const unsigned short* bh = &mb_hi[nt & 1][m][8 * hh];
        const unsigned short* bl = &mb_lo[nt & 1][m][8 * hh];

        v8f acc = {0.f, 0.f, 0.f, 0.f, 0.f, 0.f, 0.f, 0.f};
#pragma unroll
        for (int kb = 0; kb < 8; ++kb) {
            Frag Bh, Bl;
            Bh.u[0] = *(const uint4*)(bh + 32 * kb);
            Bh.u[1] = *(const uint4*)(bh + 32 * kb + 16);
            Bl.u[0] = *(const uint4*)(bl + 32 * kb);
            Bl.u[1] = *(const uint4*)(bl + 32 * kb + 16);
            acc = WMMA_BF16(Ah[kb].v, Bh.v, acc);
            acc = WMMA_BF16(Ah[kb].v, Bl.v, acc);
            acc = WMMA_BF16(Al[kb].v, Bh.v, acc);
        }
        // C layout: VGPR rr holds M = rr + 8h, lane column = N
#pragma unroll
        for (int rr = 0; rr < 8; ++rr) {
            const float v = acc[rr];
            if (v > best[rr]) { best[rr] = v; bidx[rr] = cidx; }
        }

        if (wv == 0) __builtin_amdgcn_s_wait_tensorcnt(0);
        __syncthreads();
    }

    // Cross-lane argmax over the 16 N lanes of each half (ties -> lowest index)
#pragma unroll
    for (int off = 8; off >= 1; off >>= 1) {
#pragma unroll
        for (int rr = 0; rr < 8; ++rr) {
            const float ov = __shfl_xor(best[rr], off, 32);
            const int   oi = __shfl_xor(bidx[rr], off, 32);
            if (ov > best[rr] || (ov == best[rr] && oi < bidx[rr])) {
                best[rr] = ov; bidx[rr] = oi;
            }
        }
    }

    // ---- Phase 3: scatter-add (all 32 lanes cooperate per row) ----
#pragma unroll
    for (int mm = 0; mm < 16; ++mm) {
        const int bkt = __shfl(bidx[mm & 7], (mm >> 3) * 16, 32);
        if (lane == mm) atomicAdd(&bins[bkt], 1.0f);
        const int d0 = lane * 8;
        float* dst = sums + (size_t)bkt * 256 + d0;
        const unsigned short* ph = &sh_hi[wv][mm][d0];
        const unsigned short* pl = &sh_lo[wv][mm][d0];
#pragma unroll
        for (int j = 0; j < 8; ++j) {
            atomicAdd(dst + j, bf16_to_f32(ph[j]) + bf16_to_f32(pl[j]));
        }
    }
}

// ---------------------------------------------------------------------------
// Kernel 3: x_global = bins ? l2norm(sums) : means_n; write fp32 out + hi/lo
// ---------------------------------------------------------------------------
__global__ void __launch_bounds__(256)
finalize_kernel(const float* __restrict__ sums,
                const float* __restrict__ bins,
                const float* __restrict__ means_n,
                float* __restrict__ out_xg,
                unsigned short* __restrict__ xh,
                unsigned short* __restrict__ xl)
{
    const int c = blockIdx.x;
    const int t = threadIdx.x;
    const float s = sums[c * 256 + t];
    float ss = s * s;
#pragma unroll
    for (int off = 16; off >= 1; off >>= 1) ss += __shfl_xor(ss, off, 32);
    __shared__ float red[8];
    if ((t & 31) == 0) red[t >> 5] = ss;
    __syncthreads();
    float tot = 0.f;
#pragma unroll
    for (int i = 0; i < 8; ++i) tot += red[i];
    const float inv = 1.0f / fmaxf(sqrtf(tot), 1e-12f);
    const float xn = (bins[c] != 0.0f) ? s * inv : means_n[c * 256 + t];
    out_xg[c * 256 + t] = xn;
    const unsigned short h = f32_to_bf16(xn);
    xh[c * 256 + t] = h;
    xl[c * 256 + t] = f32_to_bf16(xn - bf16_to_f32(h));
}

// ---------------------------------------------------------------------------
// Kernel 4: k = xg @ Wk^T, v = xg @ Wv^T via split-bf16 WMMA;
// store with [c, 8, 32] -> [8, c, 32] head transpose folded in.
// ---------------------------------------------------------------------------
__global__ void __launch_bounds__(256)
irca_kv_kernel(const unsigned short* __restrict__ xg_h,
               const unsigned short* __restrict__ xg_l,
               const unsigned short* __restrict__ wk_h,
               const unsigned short* __restrict__ wk_l,
               const unsigned short* __restrict__ wv_h,
               const unsigned short* __restrict__ wv_l,
               float* __restrict__ out)
{
    const int lane = threadIdx.x & 31;
    const int w    = blockIdx.x * 8 + (threadIdx.x >> 5);
    const int gemm = w >> 8;
    const int tid  = w & 255;
    const int tm = tid >> 4, tn = tid & 15;
    const unsigned short* bhp = gemm ? wv_h : wk_h;
    const unsigned short* blp = gemm ? wv_l : wk_l;
    float* o = out + gemm * 65536;

    const int m = lane & 15, hh = lane >> 4;
    const unsigned short* arow_h = xg_h + (size_t)(tm * 16 + m) * 256;
    const unsigned short* arow_l = xg_l + (size_t)(tm * 16 + m) * 256;
    const int n = tn * 16 + m;
    const unsigned short* bh = bhp + (size_t)n * 256 + 8 * hh;
    const unsigned short* bl = blp + (size_t)n * 256 + 8 * hh;

    v8f acc = {0.f, 0.f, 0.f, 0.f, 0.f, 0.f, 0.f, 0.f};
#pragma unroll
    for (int kb = 0; kb < 8; ++kb) {
        Frag Ah, Al, Bh, Bl;
        const int c0 = 32 * kb + 8 * hh;
        Ah.u[0] = *(const uint4*)(arow_h + c0);
        Ah.u[1] = *(const uint4*)(arow_h + c0 + 16);
        Al.u[0] = *(const uint4*)(arow_l + c0);
        Al.u[1] = *(const uint4*)(arow_l + c0 + 16);
        Bh.u[0] = *(const uint4*)(bh + 32 * kb);
        Bh.u[1] = *(const uint4*)(bh + 32 * kb + 16);
        Bl.u[0] = *(const uint4*)(bl + 32 * kb);
        Bl.u[1] = *(const uint4*)(bl + 32 * kb + 16);
        acc = WMMA_BF16(Ah.v, Bh.v, acc);
        acc = WMMA_BF16(Ah.v, Bl.v, acc);
        acc = WMMA_BF16(Al.v, Bh.v, acc);
    }
#pragma unroll
    for (int rr = 0; rr < 8; ++rr) {
        const int mrow = tm * 16 + rr + 8 * hh;   // cluster index
        const int q    = tn * 16 + (lane & 15);   // feature index
        const int head = q >> 5, dh = q & 31;
        o[head * 8192 + mrow * 32 + dh] = acc[rr];
    }
}

// ---------------------------------------------------------------------------
// Host launch
// ---------------------------------------------------------------------------
extern "C" void kernel_launch(void* const* d_in, const int* in_sizes, int n_in,
                              void* d_out, int out_size, void* d_ws, size_t ws_size,
                              hipStream_t stream)
{
    const float* x  = (const float*)d_in[0];
    const float* xm = (const float*)d_in[1];
    const float* Wk = (const float*)d_in[2];
    const float* Wv = (const float*)d_in[3];
    float* out = (float*)d_out;

    char* ws = (char*)d_ws;
    float*          means_n = (float*)(ws + 0);               // 256 KB
    unsigned short* mh      = (unsigned short*)(ws + 262144); // 128 KB
    unsigned short* ml_     = (unsigned short*)(ws + 393216); // 128 KB
    unsigned short* wkh     = (unsigned short*)(ws + 524288);
    unsigned short* wkl     = (unsigned short*)(ws + 655360);
    unsigned short* wvh     = (unsigned short*)(ws + 786432);
    unsigned short* wvl     = (unsigned short*)(ws + 917504);
    float*          sums    = (float*)(ws + 1048576);         // 256 KB
    float*          bins    = (float*)(ws + 1310720);         // 1 KB
    unsigned short* xgh     = (unsigned short*)(ws + 1311744);
    unsigned short* xgl     = (unsigned short*)(ws + 1442816);

    // zero accumulators every call (deterministic across graph replays)
    (void)hipMemsetAsync(sums, 0, 262144 + 1024, stream);

    row_l2norm_split_kernel<<<256, 256, 0, stream>>>(xm, means_n, mh, ml_);
    split_kernel<<<256, 256, 0, stream>>>(Wk, wkh, wkl, 65536);
    split_kernel<<<256, 256, 0, stream>>>(Wv, wvh, wvl, 65536);
    irca_dist_kernel<<<1024, 128, 0, stream>>>(x, mh, ml_, sums, bins);
    finalize_kernel<<<256, 256, 0, stream>>>(sums, bins, means_n,
                                             out + 131072, xgh, xgl);
    irca_kv_kernel<<<64, 256, 0, stream>>>(xgh, xgl, wkh, wkl, wvh, wvl, out);
}
